// InterpretableMultiHeadAttention_19061064859719
// MI455X (gfx1250) — matrix operand order repaired and checked by reading, not compiled, not run on hardware
//
#include <hip/hip_runtime.h>
#include <hip/hip_bf16.h>
#include <cstdint>
#include <cstddef>

// ---------------- problem constants ----------------
#define Bn   4
#define Tn   2048
#define HIDn 1024
#define NHn  16
#define HSn  64
#define MTOT (Bn * Tn)            // 8192 rows total
#define ATTN_SCALE 0.125f         // HS^-0.5

// Make an LDS object visible as "written by something the compiler can't see"
// (the Tensor Data Mover). Escapes the pointer + memory clobber.
#define LDS_ESCAPE(p) asm volatile("" : : "r"(p) : "memory")

typedef __attribute__((ext_vector_type(16))) __bf16 v16bf;
typedef __attribute__((ext_vector_type(8)))  __bf16 v8bf;
typedef __attribute__((ext_vector_type(8)))  float  v8f;
typedef __attribute__((ext_vector_type(2)))  float  v2f;
typedef __attribute__((ext_vector_type(4)))  uint32_t v4u;
typedef __attribute__((ext_vector_type(8)))  int      v8i;
typedef __attribute__((ext_vector_type(4)))  int      v4i;

__device__ __forceinline__ __bf16 f2bf(float f) {
  unsigned u = __float_as_uint(f);
  unsigned r = (u + 0x7FFFu + ((u >> 16) & 1u)) >> 16;  // round-to-nearest-even
  unsigned short s = (unsigned short)r;
  __bf16 out;
  __builtin_memcpy(&out, &s, sizeof(out));
  return out;
}

// A fragment 16x32 bf16 from row-major source (leading dim ld).
// Order used here (not ISA table order): lane m=lane&15, half=lane>>4; elems 0..15 -> k=half*16+e (contiguous, matches B).
__device__ __forceinline__ v16bf load_a_bf16(const __bf16* __restrict__ p, int ld, int lane) {
  const int m = lane & 15, h = lane >> 4;
  const __bf16* row = p + (size_t)m * ld + h * 16;
  v8bf lo = *(const v8bf*)(row);
  v8bf hi = *(const v8bf*)(row + 8);
  v16bf a;
#pragma unroll
  for (int i = 0; i < 8; ++i) { a[i] = lo[i]; a[8 + i] = hi[i]; }
  return a;
}

// Same A fragment from an fp32 source, converted to bf16 on the fly.
__device__ __forceinline__ v16bf load_a_f32(const float* __restrict__ p, int ld, int lane) {
  const int m = lane & 15, h = lane >> 4;
  const float* row = p + (size_t)m * ld + h * 16;
  v16bf a;
#pragma unroll
  for (int i = 0; i < 8; ++i) { a[i] = f2bf(row[i]); a[8 + i] = f2bf(row[8 + i]); }
  return a;
}

// B fragment 32x16 bf16 where element (k,n) = p[n*ld + k] (transposed source ->
// per-lane k = 16*(lane>>4)+e is contiguous; two 16B loads). Works for global or LDS.
__device__ __forceinline__ v16bf load_bt_bf16(const __bf16* p, int ld, int lane) {
  const int n = lane & 15, kh = (lane >> 4) * 16;
  const __bf16* row = p + (size_t)n * ld + kh;
  v8bf lo = *(const v8bf*)(row);
  v8bf hi = *(const v8bf*)(row + 8);
  v16bf b;
#pragma unroll
  for (int i = 0; i < 8; ++i) { b[i] = lo[i]; b[8 + i] = hi[i]; }
  return b;
}

// B fragment 32x16 bf16 where element (k,n) = p[k*ld + n] (strided; small V only).
__device__ __forceinline__ v16bf load_b_bf16(const __bf16* __restrict__ p, int ld, int lane) {
  const int n = lane & 15, kh = (lane >> 4) * 16;
  v16bf b;
#pragma unroll
  for (int e = 0; e < 16; ++e) b[e] = p[(size_t)(kh + e) * ld + n];
  return b;
}

__device__ __forceinline__ v8f wmma_bf16(v16bf a, v16bf b, v8f c) {
  return __builtin_amdgcn_wmma_f32_16x16x32_bf16(false, a, false, b, (short)0, c, false, false);
}

// ---- Tensor Data Mover: 2-D bf16 tile (tile_d0 contiguous elems x tile_d1 rows)
// from global (row stride = stride0 elems) into LDS at byte offset lds_off.
// D# bit layout per CDNA5 ISA ch.8 (group0: count/lds/global/type, group1: dims).
// This toolchain's builtin takes 6 args (g0, g1, g2, g3, g4, cpol).
__device__ __forceinline__ void tdm_load_2d_bf16(
    uint32_t lds_off, const __bf16* gsrc,
    uint32_t tensor_d0, uint32_t tensor_d1, uint32_t stride0,
    uint32_t tile_d0, uint32_t tile_d1) {
  uint64_t ga = (uint64_t)(uintptr_t)gsrc;
  v4u g0;
  g0.x = 1u;                                   // count=1, user descriptor
  g0.y = lds_off;                              // lds_addr (bytes, WG-relative)
  g0.z = (uint32_t)ga;                         // global_addr[31:0]
  g0.w = (uint32_t)(ga >> 32) | (2u << 30);    // global_addr[56:32] | type=2
  v8i g1;
  g1[0] = (int)(1u << 16);                     // data_size=1 (2B), wg_mask=0
  g1[1] = (int)((tensor_d0 & 0xFFFFu) << 16);  // tensor_dim0[15:0] @bits63:48
  g1[2] = (int)((tensor_d0 >> 16) | ((tensor_d1 & 0xFFFFu) << 16));
  g1[3] = (int)((tensor_d1 >> 16) | (tile_d0 << 16));   // tile_dim0 @bits127:112
  g1[4] = (int)(tile_d1 & 0xFFFFu);            // tile_dim1; tile_dim2=0
  g1[5] = (int)stride0;                        // tensor_dim0_stride[31:0]
  g1[6] = 0;                                   // stride0[47:32]=0, stride1 lo=0
  g1[7] = 0;
  v4i z4 = {0, 0, 0, 0};
  v8i z8 = {0, 0, 0, 0, 0, 0, 0, 0};
  __builtin_amdgcn_tensor_load_to_lds(g0, g1, z4, z4, z8, 0);
}

// ---------------- packing kernels ----------------
__global__ __launch_bounds__(256) void cvt_bf16_kernel(const float* __restrict__ in,
                                                       __bf16* __restrict__ out, int nElem) {
  int i = blockIdx.x * 256 + threadIdx.x;
  if (i < nElem) out[i] = f2bf(in[i]);
}

// in: rows x cols fp32 -> out: cols x rows bf16 (transpose)
__global__ __launch_bounds__(256) void cvt_transpose_bf16_kernel(
    const float* __restrict__ in, __bf16* __restrict__ out, int rows, int cols) {
  int i = blockIdx.x * 256 + threadIdx.x;
  if (i >= rows * cols) return;
  int r = i / cols, c = i - r * cols;
  out[(size_t)c * rows + r] = f2bf(in[i]);
}

// ---------------- projection GEMM with TDM-staged weights --------------------
// Out = X*W + bias. X: MxK bf16, Wt: NxK bf16 (transposed weights).
// WG = 8 waves = 256x64 output block; each wave computes 32x64 (8 accumulators).
// The 64x64 bf16 weight chunk (shared by all 8 waves) is DMA'd into LDS by the
// Tensor Data Mover, double-buffered, tracked with TENSORcnt. A-rows are
// per-wave and stream straight from global (L2-resident).
#define PROJ_MB 256
#define PROJ_NB 64
#define PROJ_KC 64
__global__ __launch_bounds__(256) void proj_gemm_kernel(
    const __bf16* __restrict__ X, const __bf16* __restrict__ Wt,
    const float* __restrict__ bias, __bf16* __restrict__ Out,
    int M, int N, int K) {
  // NOTE: single __shared__ object in this kernel -> LDS byte offset 0.
  __shared__ __bf16 bsm[2][PROJ_NB * PROJ_KC];   // 2 x 8 KB
  LDS_ESCAPE(&bsm[0][0]);   // TDM writes here behind the compiler's back

  const int wave = threadIdx.x >> 5, lane = threadIdx.x & 31;
  const int nb = N / PROJ_NB;
  const int m0 = (blockIdx.x / nb) * PROJ_MB + wave * 32;
  const int n0 = (blockIdx.x % nb) * PROJ_NB;
  const int NC = K / PROJ_KC;

  v8f acc[2][4];
#pragma unroll
  for (int i = 0; i < 2; ++i)
#pragma unroll
    for (int j = 0; j < 4; ++j) acc[i][j] = (v8f){};

  const __bf16* Wbase = Wt + (size_t)n0 * K;
  if (threadIdx.x < 32)   // wave 0 drives the TDM (EXEC ignored by tensor ops)
    tdm_load_2d_bf16(0u, Wbase, (uint32_t)K, (uint32_t)PROJ_NB, (uint32_t)K,
                     PROJ_KC, PROJ_NB);

  for (int kc = 0; kc < NC; ++kc) {
    if (threadIdx.x < 32) {
      if (kc + 1 < NC) {
        tdm_load_2d_bf16((uint32_t)(((kc + 1) & 1) * PROJ_NB * PROJ_KC * 2),
                         Wbase + (kc + 1) * PROJ_KC, (uint32_t)K,
                         (uint32_t)PROJ_NB, (uint32_t)K, PROJ_KC, PROJ_NB);
        __builtin_amdgcn_s_wait_tensorcnt(1);   // current chunk landed
      } else {
        __builtin_amdgcn_s_wait_tensorcnt(0);
      }
    }
    __syncthreads();

    const __bf16* bc = &bsm[kc & 1][0];
    LDS_ESCAPE(bc);   // clobber point: re-materialize fragment loads from LDS
    const __bf16* Ap = X + (size_t)m0 * K + kc * PROJ_KC;
#pragma unroll
    for (int ks = 0; ks < 2; ++ks) {
      v16bf a0 = load_a_bf16(Ap + ks * 32,          K, lane);
      v16bf a1 = load_a_bf16(Ap + (size_t)16 * K + ks * 32, K, lane);
      v16bf bf[4];
#pragma unroll
      for (int j = 0; j < 4; ++j)
        bf[j] = load_bt_bf16(bc + (j * 16) * PROJ_KC + ks * 32, PROJ_KC, lane);
#pragma unroll
      for (int j = 0; j < 4; ++j) {
        acc[0][j] = wmma_bf16(a0, bf[j], acc[0][j]);
        acc[1][j] = wmma_bf16(a1, bf[j], acc[1][j]);
      }
    }
    __syncthreads();   // buffer kc&1 is free for reuse at chunk kc+2
  }

  const int n = lane & 15, mh = (lane >> 4) * 8;
#pragma unroll
  for (int i = 0; i < 2; ++i)
#pragma unroll
    for (int j = 0; j < 4; ++j) {
      const float bvv = bias[n0 + j * 16 + n];
#pragma unroll
      for (int r = 0; r < 8; ++r)
        Out[(size_t)(m0 + 16 * i + mh + r) * N + n0 + j * 16 + n] =
            f2bf(acc[i][j][r] + bvv);
    }
}

// ---------------- scores + softmax + head-average -----------------------
// One WG (512 thr = 16 waves) per (batch, 16-query-row strip). Each wave owns
// 8 of the 128 key tiles; the 16x2048 score strip for one head lives in
// registers. shfl_xor within 16-lane groups + 1KB LDS cross-wave reduce.
__global__ __launch_bounds__(512) void attn_softmax_kernel(
    const __bf16* __restrict__ Qb, const __bf16* __restrict__ Kb,
    float* __restrict__ attn) {
  const int b    = blockIdx.x >> 7;
  const int q0   = (blockIdx.x & 127) << 4;
  const int wave = threadIdx.x >> 5;
  const int lane = threadIdx.x & 31;
  const int half = lane >> 4;
  const int n    = lane & 15;

  __shared__ float red[16][16];              // [row][wave]

  const __bf16* Qbase = Qb + (size_t)(b * Tn + q0) * HIDn;
  const __bf16* Kbase = Kb + (size_t)(b * Tn) * HIDn;

  float pacc[8][8];
#pragma unroll
  for (int t = 0; t < 8; ++t)
#pragma unroll
    for (int r = 0; r < 8; ++r) pacc[t][r] = 0.f;

  for (int h = 0; h < NHn; ++h) {
    v16bf aq0 = load_a_bf16(Qbase + h * HSn,      HIDn, lane);
    v16bf aq1 = load_a_bf16(Qbase + h * HSn + 32, HIDn, lane);
    v8f s[8];
#pragma unroll
    for (int t = 0; t < 8; ++t) {
      const int kt0 = (wave * 8 + t) << 4;
      const __bf16* kp = Kbase + (size_t)kt0 * HIDn + h * HSn;
      v16bf b0 = load_bt_bf16(kp,      HIDn, lane);
      v16bf b1 = load_bt_bf16(kp + 32, HIDn, lane);
      v8f c = {};
      c = wmma_bf16(aq0, b0, c);
      c = wmma_bf16(aq1, b1, c);
      s[t] = c * ATTN_SCALE;
    }
    float pm[8];
#pragma unroll
    for (int r = 0; r < 8; ++r) {
      float v = s[0][r];
#pragma unroll
      for (int t = 1; t < 8; ++t) v = fmaxf(v, s[t][r]);
      pm[r] = v;
    }
#pragma unroll
    for (int off = 8; off >= 1; off >>= 1)
#pragma unroll
      for (int r = 0; r < 8; ++r) pm[r] = fmaxf(pm[r], __shfl_xor(pm[r], off, 32));
    if (n == 0)
#pragma unroll
      for (int r = 0; r < 8; ++r) red[half * 8 + r][wave] = pm[r];
    __syncthreads();
    float fm[8];
#pragma unroll
    for (int r = 0; r < 8; ++r) {
      float v = red[half * 8 + r][0];
      for (int w = 1; w < 16; ++w) v = fmaxf(v, red[half * 8 + r][w]);
      fm[r] = v;
    }
    __syncthreads();
    float ps[8];
#pragma unroll
    for (int r = 0; r < 8; ++r) ps[r] = 0.f;
#pragma unroll
    for (int t = 0; t < 8; ++t)
#pragma unroll
      for (int r = 0; r < 8; ++r) {
        float e = __expf(s[t][r] - fm[r]);
        s[t][r] = e;
        ps[r] += e;
      }
#pragma unroll
    for (int off = 8; off >= 1; off >>= 1)
#pragma unroll
      for (int r = 0; r < 8; ++r) ps[r] += __shfl_xor(ps[r], off, 32);
    if (n == 0)
#pragma unroll
      for (int r = 0; r < 8; ++r) red[half * 8 + r][wave] = ps[r];
    __syncthreads();
    float winv[8];
#pragma unroll
    for (int r = 0; r < 8; ++r) {
      float v = 0.f;
      for (int w = 0; w < 16; ++w) v += red[half * 8 + r][w];
      winv[r] = 1.0f / (v * (float)NHn);     // normalize + head-mean
    }
    __syncthreads();
#pragma unroll
    for (int t = 0; t < 8; ++t)
#pragma unroll
      for (int r = 0; r < 8; ++r) pacc[t][r] += s[t][r] * winv[r];
  }

  float* abase = attn + (size_t)b * Tn * Tn;
#pragma unroll
  for (int t = 0; t < 8; ++t) {
    const int kt0 = (wave * 8 + t) << 4;
#pragma unroll
    for (int r = 0; r < 8; ++r)
      abase[(size_t)(q0 + half * 8 + r) * Tn + kt0 + n] = pacc[t][r];
  }
}

// ---------------- O1 = attn_avg @ V (fp32 attn -> bf16 on the fly) ----------
__global__ __launch_bounds__(256) void attnv_kernel(
    const float* __restrict__ attn, const __bf16* __restrict__ Vb,
    float* __restrict__ O1) {
  const int wave = threadIdx.x >> 5, lane = threadIdx.x & 31;
  const int gw = blockIdx.x * 8 + wave;       // 4*128*4 = 2048 tiles
  const int mt = gw >> 2, nt = gw & 3;
  const int b  = mt >> 7;
  const int m0 = (mt & 127) << 4;
  const int n0 = nt << 4;
  const float*  A  = attn + (size_t)b * Tn * Tn + (size_t)m0 * Tn;
  const __bf16* Bv = Vb   + (size_t)b * Tn * HSn + n0;
  v8f c = {};
  for (int k0 = 0; k0 < Tn; k0 += 32) {
    v16bf a  = load_a_f32(A + k0, Tn, lane);
    v16bf bb = load_b_bf16(Bv + (size_t)k0 * HSn, HSn, lane);
    c = wmma_bf16(a, bb, c);
  }
  const int n = lane & 15, mh = (lane >> 4) * 8;
#pragma unroll
  for (int r = 0; r < 8; ++r)
    O1[(size_t)(b * Tn + m0 + mh + r) * HSn + n0 + n] = c[r];
}

// ---------------- out = O1 @ Wo + bo via fp32 WMMA (16x16x4) ---------------
// Keeps the final projection in full fp32 precision on the matrix pipe.
// A 16x4 fp32: lane m=lane&15, k=half*2+{0,1}. B 4x16: lane n=lane&15, same k.
__global__ __launch_bounds__(256) void out_proj_kernel(
    const float* __restrict__ O1, const float* __restrict__ Wo,
    const float* __restrict__ bo, float* __restrict__ out) {
  const int wave = threadIdx.x >> 5, lane = threadIdx.x & 31;
  const int gw = blockIdx.x * 8 + wave;       // (8192/16)*(1024/16) tiles
  const int m0 = (gw >> 6) << 4;
  const int n0 = (gw & 63) << 4;
  const int n = lane & 15, half = lane >> 4, m = lane & 15;
  const float* Ap = O1 + (size_t)m0 * HSn + m * HSn + half * 2;
  const float* Bp = Wo + (size_t)(half * 2) * HIDn + n0 + n;
  v8f c = {};
#pragma unroll
  for (int k0 = 0; k0 < HSn; k0 += 4) {
    v2f a, b;
    a[0] = Ap[k0];
    a[1] = Ap[k0 + 1];
    b[0] = Bp[(size_t)k0 * HIDn];
    b[1] = Bp[(size_t)(k0 + 1) * HIDn];
    c = __builtin_amdgcn_wmma_f32_16x16x4_f32(false, a, false, b, (short)0, c,
                                              false, false);
  }
  const float bvv = bo[n0 + n];
#pragma unroll
  for (int r = 0; r < 8; ++r)
    out[(size_t)(m0 + half * 8 + r) * HIDn + n0 + n] = c[r] + bvv;
}

// ---------------- host launch ----------------
extern "C" void kernel_launch(void* const* d_in, const int* in_sizes, int n_in,
                              void* d_out, int out_size, void* d_ws, size_t ws_size,
                              hipStream_t stream) {
  (void)in_sizes; (void)n_in; (void)out_size; (void)ws_size;
  const float* x  = (const float*)d_in[0];
  const float* Wq = (const float*)d_in[1];
  const float* bq = (const float*)d_in[2];
  const float* Wk = (const float*)d_in[3];
  const float* bk = (const float*)d_in[4];
  const float* Wv = (const float*)d_in[5];
  const float* bv = (const float*)d_in[6];
  const float* Wo = (const float*)d_in[7];
  const float* bo = (const float*)d_in[8];

  float* out  = (float*)d_out;                         // (B,T,HID)
  float* attn = out + (size_t)MTOT * HIDn;             // (B,T,T)

  // workspace layout (~58 MB total)
  char* w = (char*)d_ws;
  __bf16* xb  = (__bf16*)w; w += (size_t)MTOT * HIDn * 2;
  __bf16* Wqt = (__bf16*)w; w += (size_t)HIDn * HIDn * 2;
  __bf16* Wkt = (__bf16*)w; w += (size_t)HIDn * HIDn * 2;
  __bf16* Wvt = (__bf16*)w; w += (size_t)HIDn * HSn  * 2;
  __bf16* Qb  = (__bf16*)w; w += (size_t)MTOT * HIDn * 2;
  __bf16* Kb  = (__bf16*)w; w += (size_t)MTOT * HIDn * 2;
  __bf16* Vb  = (__bf16*)w; w += (size_t)MTOT * HSn  * 2;
  float*  O1  = (float*)w;  w += (size_t)MTOT * HSn  * 4;

  // 1) fp32 -> bf16 packing (weights transposed for contiguous B fragments)
  cvt_bf16_kernel<<<(MTOT * HIDn) / 256, 256, 0, stream>>>(x, xb, MTOT * HIDn);
  cvt_transpose_bf16_kernel<<<(HIDn * HIDn) / 256, 256, 0, stream>>>(Wq, Wqt, HIDn, HIDn);
  cvt_transpose_bf16_kernel<<<(HIDn * HIDn) / 256, 256, 0, stream>>>(Wk, Wkt, HIDn, HIDn);
  cvt_transpose_bf16_kernel<<<(HIDn * HSn)  / 256, 256, 0, stream>>>(Wv, Wvt, HIDn, HSn);

  // 2) projections (WMMA bf16 + TDM-staged weight tiles, fused bias)
  proj_gemm_kernel<<<(MTOT / PROJ_MB) * (HIDn / PROJ_NB), 256, 0, stream>>>(
      xb, Wqt, bq, Qb, MTOT, HIDn, HIDn);
  proj_gemm_kernel<<<(MTOT / PROJ_MB) * (HIDn / PROJ_NB), 256, 0, stream>>>(
      xb, Wkt, bk, Kb, MTOT, HIDn, HIDn);
  proj_gemm_kernel<<<(MTOT / PROJ_MB) * (HSn / PROJ_NB), 256, 0, stream>>>(
      xb, Wvt, bv, Vb, MTOT, HSn, HIDn);

  // 3) QK^T scores (WMMA) + fp32 softmax + head-average -> attn_avg output
  attn_softmax_kernel<<<Bn * (Tn / 16), 512, 0, stream>>>(Qb, Kb, attn);

  // 4) O1 = attn_avg @ V  (WMMA)
  attnv_kernel<<<(Bn * (Tn / 16) * (HSn / 16)) / 8, 256, 0, stream>>>(attn, Vb, O1);

  // 5) out = O1 @ Wo + bo (fp32 WMMA 16x16x4)
  out_proj_kernel<<<(MTOT / 16) * (HIDn / 16) / 8, 256, 0, stream>>>(O1, Wo, bo, out);
}
